// DifferentialAttention_84679575208071
// MI455X (gfx1250) — matrix operand order repaired and checked by reading, not compiled, not run on hardware
//
#include <hip/hip_runtime.h>
#include <math.h>

// ---------------------------------------------------------------------------
// Differential attention, MI455X (gfx1250, wave32, WMMA f32_16x16x32_f16)
// ---------------------------------------------------------------------------
typedef _Float16 half_t;
typedef __attribute__((ext_vector_type(16))) _Float16 v16h;
typedef __attribute__((ext_vector_type(8)))  _Float16 v8h;
typedef __attribute__((ext_vector_type(8)))  float    v8f;

#define B_  2
#define S_  2048
#define D_  1024
#define H_  16
#define HD_ 64
#define THD_ 1024

// ---- WMMA fragment helpers (CDNA5 16x16x32 f16 layouts) -------------------
// A (16x32, f16): lane L holds row M=L&15; halves: lanes<16 -> K=0..7,16..23,
// lanes>=16 -> K=8..15,24..31.  Two 128-bit loads per lane.
__device__ __forceinline__ v16h frag_a_load(const half_t* base, int ld) {
  const int lane = threadIdx.x & 31;
  const half_t* p = base + (lane & 15) * ld + ((lane & 16) ? 8 : 0);
  v8h lo = *(const v8h*)(p);
  v8h hi = *(const v8h*)(p + 16);
  v16h f;
#pragma unroll
  for (int i = 0; i < 8; ++i) { f[i] = lo[i]; f[i + 8] = hi[i]; }
  return f;
}

// B (32x16, f16) read from a [N][K] (i.e. B-transposed, K contiguous) tile:
// lane L holds column N=L&15; halves: lanes<16 -> K=0..7,16..23, lanes>=16 -> K=8..15,24..31.
__device__ __forceinline__ v16h frag_b_load(const half_t* base, int ld) {
  const int lane = threadIdx.x & 31;
  const half_t* p = base + (lane & 15) * ld + ((lane & 16) ? 8 : 0);
  v8h lo = *(const v8h*)(p);
  v8h hi = *(const v8h*)(p + 16);
  v16h f;
#pragma unroll
  for (int i = 0; i < 8; ++i) { f[i] = lo[i]; f[i + 8] = hi[i]; }
  return f;
}

__device__ __forceinline__ v8f wmma16(v16h a, v16h b, v8f c) {
  return __builtin_amdgcn_wmma_f32_16x16x32_f16(false, a, false, b,
                                                (short)0, c, false, false);
}

// ---- f32 -> f16 conversion ------------------------------------------------
__global__ void cvt_f16_kernel(const float* __restrict__ in,
                               half_t* __restrict__ out, int n) {
  int i = blockIdx.x * blockDim.x + threadIdx.x;
  if (i < n) out[i] = (half_t)in[i];
}

// ---- generic 128x128x32 WMMA GEMM, double-buffered LDS pipeline -----------
// C = A(f16,MxK) * B(f16,KxN).  OUTF16: 0 -> f32 output, 1 -> f16 output.
template <int OUTF16>
__global__ __launch_bounds__(256) void gemm128_kernel(
    const half_t* __restrict__ A, const half_t* __restrict__ Bm,
    void* __restrict__ Cv, int M, int N, int K) {
  constexpr int LD = 40;  // padded K-stride (halves); 80B rows, 16B aligned
  __shared__ __align__(16) half_t sa[2][128 * LD];   // A tiles [m][k]
  __shared__ __align__(16) half_t sbt[2][128 * LD];  // B tiles transposed [n][k]
  const int tid = threadIdx.x;
  const int wid = tid >> 5;
  const int m0 = blockIdx.y * 128;
  const int n0 = blockIdx.x * 128;
  const int wm = (wid & 3) * 32;   // 4 waves along M (32 rows each)
  const int wn = (wid >> 2) * 64;  // 2 waves along N (64 cols each)

  // per-thread staging coordinates
  const int a_row0 = tid >> 2,      a_c8 = (tid & 3) * 8;         // seg i=0
  const int a_row1 = (256 + tid) >> 2, a_c81 = ((256 + tid) & 3) * 8;
  const int b_kr0 = tid >> 4,       b_c8 = (tid & 15) * 8;
  const int b_kr1 = (256 + tid) >> 4,  b_c81 = ((256 + tid) & 15) * 8;

  v8f acc[2][4];
#pragma unroll
  for (int i = 0; i < 2; ++i)
#pragma unroll
    for (int j = 0; j < 4; ++j)
#pragma unroll
      for (int g = 0; g < 8; ++g) acc[i][j][g] = 0.0f;

  v8h ra0, ra1, rb0, rb1;
  // prologue: fetch k-slice 0
  ra0 = *(const v8h*)(A + (size_t)(m0 + a_row0) * K + 0 + a_c8);
  ra1 = *(const v8h*)(A + (size_t)(m0 + a_row1) * K + 0 + a_c81);
  rb0 = *(const v8h*)(Bm + (size_t)(0 + b_kr0) * N + n0 + b_c8);
  rb1 = *(const v8h*)(Bm + (size_t)(0 + b_kr1) * N + n0 + b_c81);
  {
    *(v8h*)(sa[0] + a_row0 * LD + a_c8) = ra0;
    *(v8h*)(sa[0] + a_row1 * LD + a_c81) = ra1;
#pragma unroll
    for (int e = 0; e < 8; ++e) sbt[0][(b_c8 + e) * LD + b_kr0] = rb0[e];
#pragma unroll
    for (int e = 0; e < 8; ++e) sbt[0][(b_c81 + e) * LD + b_kr1] = rb1[e];
  }
  __syncthreads();

  int cur = 0;
  for (int kk = 0; kk < K; kk += 32) {
    const bool more = (kk + 32) < K;
    if (more) {  // issue next-slice global loads; consumed after compute
      int kn = kk + 32;
      ra0 = *(const v8h*)(A + (size_t)(m0 + a_row0) * K + kn + a_c8);
      ra1 = *(const v8h*)(A + (size_t)(m0 + a_row1) * K + kn + a_c81);
      rb0 = *(const v8h*)(Bm + (size_t)(kn + b_kr0) * N + n0 + b_c8);
      rb1 = *(const v8h*)(Bm + (size_t)(kn + b_kr1) * N + n0 + b_c81);
    }
    // compute current slice from LDS (overlaps with in-flight loads)
    v16h af[2];
#pragma unroll
    for (int mf = 0; mf < 2; ++mf)
      af[mf] = frag_a_load(sa[cur] + (wm + mf * 16) * LD, LD);
#pragma unroll
    for (int nf = 0; nf < 4; ++nf) {
      v16h bf = frag_b_load(sbt[cur] + (wn + nf * 16) * LD, LD);
#pragma unroll
      for (int mf = 0; mf < 2; ++mf)
        acc[mf][nf] = wmma16(af[mf], bf, acc[mf][nf]);
    }
    if (more) {  // store staged regs into the other buffer
      int nxt = cur ^ 1;
      *(v8h*)(sa[nxt] + a_row0 * LD + a_c8) = ra0;
      *(v8h*)(sa[nxt] + a_row1 * LD + a_c81) = ra1;
#pragma unroll
      for (int e = 0; e < 8; ++e) sbt[nxt][(b_c8 + e) * LD + b_kr0] = rb0[e];
#pragma unroll
      for (int e = 0; e < 8; ++e) sbt[nxt][(b_c81 + e) * LD + b_kr1] = rb1[e];
    }
    __syncthreads();
    cur ^= 1;
  }

  const int lane = tid & 31;
  const int colb = lane & 15;
  const int rhi = (lane & 16) ? 8 : 0;
#pragma unroll
  for (int mf = 0; mf < 2; ++mf)
#pragma unroll
    for (int nf = 0; nf < 4; ++nf)
#pragma unroll
      for (int g = 0; g < 8; ++g) {
        int row = m0 + wm + mf * 16 + g + rhi;
        int col = n0 + wn + nf * 16 + colb;
        float v = acc[mf][nf][g];
        if (OUTF16)
          ((half_t*)Cv)[(size_t)row * N + col] = (half_t)v;
        else
          ((float*)Cv)[(size_t)row * N + col] = v;
      }
}

// ---- RoPE: f32 [b][s][2*THD] -> f16 [b][br][h][s][hd] ---------------------
__global__ void rope_kernel(const float* __restrict__ qf,
                            const float* __restrict__ kf,
                            half_t* __restrict__ qro,
                            half_t* __restrict__ kro) {
  int idx = blockIdx.x * blockDim.x + threadIdx.x;  // 2^22 threads
  int d  = idx & 31;
  int h  = (idx >> 5) & 15;
  int br = (idx >> 9) & 1;
  int s  = (idx >> 10) & 2047;
  int b  = idx >> 21;
  // inv_freq = 10000^(-d/32)
  float invf = __expf(-(float)d * (9.210340371976184f / 32.0f));
  float freq = (float)s * invf;
  float sn, cs;
  __sincosf(freq, &sn, &cs);
  size_t ib = (((size_t)b * S_ + s) * (2 * THD_)) + br * THD_ + h * HD_ + d;
  float q1 = qf[ib], q2 = qf[ib + 32];
  float k1 = kf[ib], k2 = kf[ib + 32];
  size_t ob = (((size_t)((b * 2 + br) * H_ + h)) * S_ + s) * HD_ + d;
  qro[ob]      = (half_t)(q1 * cs - q2 * sn);
  qro[ob + 32] = (half_t)(q2 * cs + q1 * sn);
  kro[ob]      = (half_t)(k1 * cs - k2 * sn);
  kro[ob + 32] = (half_t)(k2 * cs + k1 * sn);
}

// ---- fused flash-style differential attention -----------------------------
// grid: (S/128, B*H); block 256 (8 waves). Each wave owns 16 query rows.
__global__ __launch_bounds__(256) void diffattn_kernel(
    const half_t* __restrict__ qro, const half_t* __restrict__ kro,
    const half_t* __restrict__ vh, const float* __restrict__ lq1,
    const float* __restrict__ lk1, const float* __restrict__ lq2,
    const float* __restrict__ lk2, const float* __restrict__ lam_init_p,
    float* __restrict__ ao) {
  constexpr int VT_LD = 136;  // padded key stride (halves), 272B rows
  constexpr int PK_LD = 136;
  __shared__ __align__(16) half_t vt[HD_ * VT_LD];       // V tile^T [d][key]
  __shared__ __align__(16) half_t plds[8 * 16 * PK_LD];  // P strips [wave][q][key]

  const int tid = threadIdx.x, wid = tid >> 5, lane = tid & 31;
  const int bh = blockIdx.y;
  const int b = bh >> 4, h = bh & 15;
  const int q0 = blockIdx.x * 128 + wid * 16;  // this wave's query base
  const float scale = 0.125f;                  // Hd^-0.5

  const half_t* qb[2];
  const half_t* kb[2];
#pragma unroll
  for (int br = 0; br < 2; ++br) {
    size_t hb = ((size_t)((b * 2 + br) * H_ + h)) * S_ * HD_;
    qb[br] = qro + hb;
    kb[br] = kro + hb;
  }
  const half_t* vbase = vh + ((size_t)b * S_) * THD_ + h * HD_;

  // Q fragments kept in registers across all key tiles
  v16h qf[2][2];
#pragma unroll
  for (int br = 0; br < 2; ++br)
#pragma unroll
    for (int ks = 0; ks < 2; ++ks)
      qf[br][ks] = frag_a_load(qb[br] + (size_t)q0 * HD_ + ks * 32, HD_);

  v8f o[2][4];
  float rmax[2][8], rsum[2][8];
#pragma unroll
  for (int br = 0; br < 2; ++br) {
#pragma unroll
    for (int nt = 0; nt < 4; ++nt)
#pragma unroll
      for (int g = 0; g < 8; ++g) o[br][nt][g] = 0.0f;
#pragma unroll
    for (int g = 0; g < 8; ++g) {
      rmax[br][g] = -__builtin_inff();
      rsum[br][g] = 0.0f;
    }
  }
  half_t* pstrip = plds + wid * 16 * PK_LD;

  for (int kt = 0; kt < S_ / 128; ++kt) {
    // prefetch next tile's K/V cachelines (global_prefetch_b8)
    if (kt + 1 < S_ / 128) {
      int prow = (kt + 1) * 128 + lane * 4;
      __builtin_prefetch(kb[0] + (size_t)prow * HD_, 0, 1);
      __builtin_prefetch(kb[1] + (size_t)prow * HD_, 0, 1);
      __builtin_prefetch(vbase + (size_t)prow * THD_, 0, 1);
    }
    // cooperative transpose-load of the 128x64 V tile into vt[d][key]
#pragma unroll
    for (int i = 0; i < 4; ++i) {
      int seg = i * 256 + tid;
      int row = seg >> 3, c8 = (seg & 7) * 8;
      v8h v = *(const v8h*)(vbase + (size_t)(kt * 128 + row) * THD_ + c8);
#pragma unroll
      for (int e = 0; e < 8; ++e) vt[(c8 + e) * VT_LD + row] = v[e];
    }
    __syncthreads();

#pragma unroll
    for (int br = 0; br < 2; ++br) {
      // S strip = Q(16x64) * K^T(64x128); K rows [key][d] are exactly the
      // [N][K]-layout B tile -> direct global b128 fragment loads.
      // Double-buffered fragments so loads for nt+1 overlap wmma of nt.
      const half_t* kbb = kb[br] + (size_t)(kt * 128) * HD_;
      v8f sacc[8];
      v16h bfd[2][2];
      bfd[0][0] = frag_b_load(kbb + 0, HD_);
      bfd[0][1] = frag_b_load(kbb + 32, HD_);
#pragma unroll
      for (int nt = 0; nt < 8; ++nt) {
        const int cur = nt & 1;
        if (nt < 7) {
          bfd[cur ^ 1][0] = frag_b_load(kbb + (size_t)((nt + 1) * 16) * HD_, HD_);
          bfd[cur ^ 1][1] =
              frag_b_load(kbb + (size_t)((nt + 1) * 16) * HD_ + 32, HD_);
        }
#pragma unroll
        for (int g = 0; g < 8; ++g) sacc[nt][g] = 0.0f;
        sacc[nt] = wmma16(qf[br][0], bfd[cur][0], sacc[nt]);
        sacc[nt] = wmma16(qf[br][1], bfd[cur][1], sacc[nt]);
      }
      // online softmax: row max across this strip (cross-lane within halves)
      float mnew[8];
#pragma unroll
      for (int g = 0; g < 8; ++g) {
        float t = sacc[0][g];
#pragma unroll
        for (int nt = 1; nt < 8; ++nt) t = fmaxf(t, sacc[nt][g]);
        t = fmaxf(t, __shfl_xor(t, 8, 32));
        t = fmaxf(t, __shfl_xor(t, 4, 32));
        t = fmaxf(t, __shfl_xor(t, 2, 32));
        t = fmaxf(t, __shfl_xor(t, 1, 32));
        mnew[g] = fmaxf(rmax[br][g], t);
      }
#pragma unroll
      for (int g = 0; g < 8; ++g) {
        float alpha = __expf(scale * (rmax[br][g] - mnew[g]));
        rsum[br][g] *= alpha;
#pragma unroll
        for (int nt = 0; nt < 4; ++nt) o[br][nt][g] *= alpha;
        rmax[br][g] = mnew[g];
      }
      // P = exp(scale*(s - m)); spill to LDS in A-fragment-friendly layout
      const int colb = lane & 15;
      const int rhi = (lane & 16) ? 8 : 0;
#pragma unroll
      for (int nt = 0; nt < 8; ++nt)
#pragma unroll
        for (int g = 0; g < 8; ++g) {
          float p = __expf(scale * (sacc[nt][g] - mnew[g]));
          rsum[br][g] += p;  // per-lane partial row sum
          pstrip[(g + rhi) * PK_LD + nt * 16 + colb] = (half_t)p;
        }
      // O += P(16x128) * V(128x64)
#pragma unroll
      for (int ks2 = 0; ks2 < 4; ++ks2) {
        v16h pa = frag_a_load(pstrip + ks2 * 32, PK_LD);
#pragma unroll
        for (int nt2 = 0; nt2 < 4; ++nt2) {
          v16h vb = frag_b_load(vt + (nt2 * 16) * VT_LD + ks2 * 32, VT_LD);
          o[br][nt2] = wmma16(pa, vb, o[br][nt2]);
        }
      }
    }
    __syncthreads();
  }

  // finalize: reduce row sums across the 16-lane halves, combine branches
#pragma unroll
  for (int br = 0; br < 2; ++br)
#pragma unroll
    for (int g = 0; g < 8; ++g) {
      float s = rsum[br][g];
      s += __shfl_xor(s, 8, 32);
      s += __shfl_xor(s, 4, 32);
      s += __shfl_xor(s, 2, 32);
      s += __shfl_xor(s, 1, 32);
      rsum[br][g] = s;
    }
  float lam = __expf(lq1[h] * lk1[h]) - __expf(lq2[h] * lk2[h]) + lam_init_p[0];
  const int colb = lane & 15;
  const int rhi = (lane & 16) ? 8 : 0;
#pragma unroll
  for (int nt = 0; nt < 4; ++nt)
#pragma unroll
    for (int g = 0; g < 8; ++g) {
      int srow = q0 + g + rhi;
      float val = o[0][nt][g] / rsum[0][g] - lam * (o[1][nt][g] / rsum[1][g]);
      ao[(((size_t)b * S_ + srow) * H_ + h) * HD_ + nt * 16 + colb] = val;
    }
}

// ---- GroupNorm over Hd per (token, head), then *(1-lambda_init), to f16 ---
__global__ __launch_bounds__(256) void groupnorm_kernel(
    const float* __restrict__ ao, const float* __restrict__ gw,
    const float* __restrict__ gb, const float* __restrict__ lam_init_p,
    half_t* __restrict__ aoh) {
  const int tid = threadIdx.x, wid = tid >> 5, lane = tid & 31;
  const int p = blockIdx.x * 8 + wid;  // (token, head) pair
  const int t = p >> 4, h = p & 15;
  size_t base = ((size_t)t * H_ + h) * HD_;
  float v0 = ao[base + lane];
  float v1 = ao[base + 32 + lane];
  float s = v0 + v1, ss = v0 * v0 + v1 * v1;
#pragma unroll
  for (int off = 16; off >= 1; off >>= 1) {
    s += __shfl_xor(s, off, 32);
    ss += __shfl_xor(ss, off, 32);
  }
  float mean = s * (1.0f / 64.0f);
  float var = ss * (1.0f / 64.0f) - mean * mean;
  float rstd = rsqrtf(var + 1e-5f);
  float sc = 1.0f - lam_init_p[0];
  int f0 = h * HD_ + lane, f1 = f0 + 32;
  aoh[(size_t)t * THD_ + f0] = (half_t)(((v0 - mean) * rstd * gw[f0] + gb[f0]) * sc);
  aoh[(size_t)t * THD_ + f1] = (half_t)(((v1 - mean) * rstd * gw[f1] + gb[f1]) * sc);
}

// ---------------------------------------------------------------------------
extern "C" void kernel_launch(void* const* d_in, const int* in_sizes, int n_in,
                              void* d_out, int out_size, void* d_ws,
                              size_t ws_size, hipStream_t stream) {
  (void)in_sizes; (void)n_in; (void)out_size; (void)ws_size;
  const float* x   = (const float*)d_in[0];
  const float* Wq  = (const float*)d_in[1];
  const float* Wk  = (const float*)d_in[2];
  const float* Wv  = (const float*)d_in[3];
  const float* Wo  = (const float*)d_in[4];
  const float* lq1 = (const float*)d_in[5];
  const float* lk1 = (const float*)d_in[6];
  const float* lq2 = (const float*)d_in[7];
  const float* lk2 = (const float*)d_in[8];
  const float* li  = (const float*)d_in[9];
  const float* gw  = (const float*)d_in[10];
  const float* gb  = (const float*)d_in[11];
  float* out = (float*)d_out;

  char* ws = (char*)d_ws;
  // workspace layout (bytes); ao/aoh alias the consumed raw q/k f32 buffers
  half_t* xh   = (half_t*)(ws + 0);          //  8 MB: x f16 [4096][1024]
  half_t* Wqh  = (half_t*)(ws + 8388608);    //  4 MB
  half_t* Wkh  = (half_t*)(ws + 12582912);   //  4 MB
  half_t* Wvh  = (half_t*)(ws + 16777216);   //  2 MB
  half_t* Woh  = (half_t*)(ws + 18874368);   //  2 MB
  half_t* qro  = (half_t*)(ws + 20971520);   // 16 MB: roped q f16 [b][br][h][s][hd]
  half_t* kro  = (half_t*)(ws + 37748736);   // 16 MB
  half_t* vh16 = (half_t*)(ws + 54525952);   //  8 MB: v f16 [b][s][h][hd]
  float*  qf32 = (float*)(ws + 62914560);    // 32 MB: raw q proj f32
  float*  kf32 = (float*)(ws + 96468992);    // 32 MB: raw k proj f32
  float*  ao   = (float*)(ws + 62914560);    // 16 MB, aliases qf32 (post-rope)
  half_t* aoh  = (half_t*)(ws + 96468992);   //  8 MB, aliases kf32 (post-rope)

  // 1) f32 -> f16 conversions
  cvt_f16_kernel<<<16384, 256, 0, stream>>>(x, xh, B_ * S_ * D_);
  cvt_f16_kernel<<<8192, 256, 0, stream>>>(Wq, Wqh, D_ * 2 * THD_);
  cvt_f16_kernel<<<8192, 256, 0, stream>>>(Wk, Wkh, D_ * 2 * THD_);
  cvt_f16_kernel<<<4096, 256, 0, stream>>>(Wv, Wvh, D_ * THD_);
  cvt_f16_kernel<<<4096, 256, 0, stream>>>(Wo, Woh, THD_ * D_);

  // 2) projections (WMMA)
  gemm128_kernel<0><<<dim3(16, 32), 256, 0, stream>>>(xh, Wqh, qf32,
                                                      B_ * S_, 2 * THD_, D_);
  gemm128_kernel<0><<<dim3(16, 32), 256, 0, stream>>>(xh, Wkh, kf32,
                                                      B_ * S_, 2 * THD_, D_);
  gemm128_kernel<1><<<dim3(8, 32), 256, 0, stream>>>(xh, Wvh, vh16,
                                                     B_ * S_, THD_, D_);

  // 3) RoPE + layout for attention
  rope_kernel<<<16384, 256, 0, stream>>>(qf32, kf32, qro, kro);

  // 4) fused differential flash attention
  diffattn_kernel<<<dim3(S_ / 128, B_ * H_), 256, 0, stream>>>(
      qro, kro, vh16, lq1, lk1, lq2, lk2, li, ao);

  // 5) GroupNorm + (1-lambda_init) scale
  groupnorm_kernel<<<(B_ * S_ * H_) / 8, 256, 0, stream>>>(ao, gw, gb, li, aoh);

  // 6) output projection (WMMA) to f32 d_out
  gemm128_kernel<0><<<dim3(8, 32), 256, 0, stream>>>(aoh, Woh, out,
                                                     B_ * S_, D_, THD_);
}
